// FancyNet_65120294142190
// MI455X (gfx1250) — matrix operand order repaired and checked by reading, not compiled, not run on hardware
//
#include <hip/hip_runtime.h>
#include <hip/hip_bf16.h>

#define HID 80
#define NB 8
#define N0 65536
#define EPSBN 1e-5f

typedef __attribute__((ext_vector_type(16))) _Float16 v16h;
typedef __attribute__((ext_vector_type(8)))  _Float16 v8h;
typedef __attribute__((ext_vector_type(8)))  float    v8f;

// ---------------------------------------------------------------------------
// WMMA GEMM row-stripe kernel. A16: [M, Kpad] f16 row-major (zero padded).
// Wt16: [Ndpad, Kpad] f16 = W transposed (zero padded). One wave per 16-row
// stripe computes NT=Ndpad/16 column tiles, branch-free 128-bit fragment loads:
//   A frag half i -> K = (i>=8?16:0) + hi*8 + (i&7)  => two contiguous v8h
//   B frag half i -> K = (i>=8?16:0) + hi*8 + (i&7)  => two contiguous v8h
// Epilogue: +bias, optional relu, optional f32 residual; writes f32 C and/or
// zero-padded f16 C (feeds the next GEMM directly).
// ---------------------------------------------------------------------------
template <int NT>
__global__ __launch_bounds__(32) void fn_gemm_wmma(
    const _Float16* __restrict__ A, int lda,
    const _Float16* __restrict__ Wt, int ldw,
    const float* __restrict__ bias,
    const float* __restrict__ residual, int ldr,
    float* __restrict__ Cf, int ldc,
    _Float16* __restrict__ Ch, int ldch,
    int Kpad, int Nd, int do_relu)
{
    const int lane = threadIdx.x & 31;
    const int hi = lane >> 4, l16 = lane & 15;
    const int row0 = blockIdx.x * 16;
    const _Float16* Ar = A + (size_t)(row0 + l16) * lda;

    v8f acc[NT];
#pragma unroll
    for (int t = 0; t < NT; ++t) acc[t] = {};

    for (int kb = 0; kb < Kpad; kb += 32) {
        v8h alo = *(const v8h*)(Ar + kb + hi * 8);
        v8h ahi = *(const v8h*)(Ar + kb + 16 + hi * 8);
        v16h a = __builtin_shufflevector(alo, ahi, 0, 1, 2, 3, 4, 5, 6, 7,
                                         8, 9, 10, 11, 12, 13, 14, 15);
#pragma unroll
        for (int t = 0; t < NT; ++t) {
            const _Float16* Br = Wt + (size_t)(t * 16 + l16) * ldw + kb;
            v8h blo = *(const v8h*)(Br + hi * 8);
            v8h bhi = *(const v8h*)(Br + 16 + hi * 8);
            v16h b = __builtin_shufflevector(blo, bhi, 0, 1, 2, 3, 4, 5, 6, 7,
                                             8, 9, 10, 11, 12, 13, 14, 15);
            acc[t] = __builtin_amdgcn_wmma_f32_16x16x32_f16(
                false, a, false, b, (short)0, acc[t], false, false);
        }
    }
#pragma unroll
    for (int t = 0; t < NT; ++t) {
        int col = t * 16 + l16;
        float bv = (bias && col < Nd) ? bias[col] : 0.0f;
#pragma unroll
        for (int r = 0; r < 8; ++r) {
            int row = row0 + r + hi * 8;   // D elem r -> M = r + hi*8
            float v = acc[t][r] + bv;
            if (do_relu) v = fmaxf(v, 0.0f);
            if (residual && col < Nd) v += residual[(size_t)row * ldr + col];
            if (Cf && col < Nd) Cf[(size_t)row * ldc + col] = v;
            if (Ch) Ch[(size_t)row * ldch + col] = (_Float16)((col < Nd) ? v : 0.0f);
        }
    }
}

// f32 [M,K] -> f16 [M,Kpad], zero padded
__global__ void fn_cvt_a16(const float* __restrict__ src, int K,
                           _Float16* __restrict__ dst, int Kpad, int M)
{
    int gid = blockIdx.x * blockDim.x + threadIdx.x;
    if (gid >= M * Kpad) return;
    int r = gid / Kpad, c = gid % Kpad;
    dst[gid] = (_Float16)((c < K) ? src[(size_t)r * K + c] : 0.0f);
}

// W f32 [K,Nd] -> Wt f16 [Ndpad,Kpad] (transposed, zero padded)
__global__ void fn_cvt_wt16(const float* __restrict__ w, int K, int Nd,
                            _Float16* __restrict__ wt, int Kpad, int Ndpad)
{
    int gid = blockIdx.x * blockDim.x + threadIdx.x;
    if (gid >= Ndpad * Kpad) return;
    int n = gid / Kpad, k = gid % Kpad;
    wt[gid] = (_Float16)((k < K && n < Nd) ? w[(size_t)k * Nd + n] : 0.0f);
}

// ---------------------------------------------------------------------------
// Brute-force KNN (k=3, includes self) per instance. grid=(B, P/256), 256 thr.
// Candidate coords streamed through 48KB LDS chunks; strict '<' insertion
// reproduces top_k lower-index tie-break. Stores GLOBAL (inst-offset) indices.
// ---------------------------------------------------------------------------
#define KNN_CHUNK 4096
__global__ void fn_knn(const float* __restrict__ coord, int* __restrict__ knn, int P)
{
    __shared__ float sc[KNN_CHUNK * 3];
    const int inst = blockIdx.x;
    const float* cb = coord + (size_t)inst * P * 3;
    const int q = blockIdx.y * blockDim.x + threadIdx.x;   // always < P
    const float qx = cb[q * 3 + 0], qy = cb[q * 3 + 1], qz = cb[q * 3 + 2];
    const float sq = qx * qx + qy * qy + qz * qz;

    float d0 = 3.4e38f, d1 = 3.4e38f, d2 = 3.4e38f;
    int   i0 = 0, i1 = 0, i2 = 0;

    for (int c0 = 0; c0 < P; c0 += KNN_CHUNK) {
        int cnt = min(KNN_CHUNK, P - c0);
        __syncthreads();
        for (int i = threadIdx.x; i < cnt * 3; i += blockDim.x)
            sc[i] = cb[(size_t)c0 * 3 + i];
        __syncthreads();
        for (int jj = 0; jj < cnt; ++jj) {
            float x = sc[jj * 3 + 0], y = sc[jj * 3 + 1], z = sc[jj * 3 + 2];
            float sj  = x * x + y * y + z * z;
            float dot = qx * x + qy * y + qz * z;
            float d = sq + sj - 2.0f * dot;
            if (d < d2) {
                int j = c0 + jj;
                if (d < d1) {
                    d2 = d1; i2 = i1;
                    if (d < d0) { d1 = d0; i1 = i0; d0 = d; i0 = j; }
                    else        { d1 = d;  i1 = j; }
                } else { d2 = d; i2 = j; }
            }
        }
    }
    int base = inst * P;
    knn[(size_t)(base + q) * 3 + 0] = base + i0;
    knn[(size_t)(base + q) * 3 + 1] = base + i1;
    knn[(size_t)(base + q) * 3 + 2] = base + i2;
}

// X16[v,c] = (f16)( src[v,c] + sum_k src[knn[v][k],c] ), padded to 96 cols
__global__ void fn_gather_sum_f16(const float* __restrict__ src,
                                  const int* __restrict__ knn,
                                  _Float16* __restrict__ dst, int n)
{
    int gid = blockIdx.x * blockDim.x + threadIdx.x;
    if (gid >= n * 96) return;
    int v = gid / 96, c = gid % 96;
    float s = 0.0f;
    if (c < HID) {
        const int* kp = knn + (size_t)v * 3;
        s = src[(size_t)v * HID + c] + src[(size_t)kp[0] * HID + c]
          + src[(size_t)kp[1] * HID + c] + src[(size_t)kp[2] * HID + c];
    }
    dst[gid] = (_Float16)s;
}

// dst[v] = sum_k src[knn[v][k]]   (f32, 80 cols) -- push-down step
__global__ void fn_gather_sum(const float* __restrict__ src, const int* __restrict__ knn,
                              float* __restrict__ dst, int n)
{
    int gid = blockIdx.x * blockDim.x + threadIdx.x;
    if (gid >= n * HID) return;
    int v = gid / HID, c = gid % HID;
    const int* kp = knn + (size_t)v * 3;
    dst[gid] = src[(size_t)kp[0] * HID + c] + src[(size_t)kp[1] * HID + c]
             + src[(size_t)kp[2] * HID + c];
}

// Per-column sum / sumsq over n rows. grid = HID blocks, 256 threads.
__global__ void fn_bn_stats(const float* __restrict__ x, float* __restrict__ stats, int n)
{
    __shared__ float ssum[256], ssq[256];
    int c = blockIdx.x, tid = threadIdx.x;
    float s = 0.0f, q = 0.0f;
    for (int r = tid; r < n; r += 256) {
        float v = x[(size_t)r * HID + c];
        s += v; q += v * v;
    }
    ssum[tid] = s; ssq[tid] = q;
    __syncthreads();
    for (int st = 128; st > 0; st >>= 1) {
        if (tid < st) { ssum[tid] += ssum[tid + st]; ssq[tid] += ssq[tid + st]; }
        __syncthreads();
    }
    if (tid == 0) {
        float m = ssum[0] / (float)n;
        stats[2 * c]     = m;
        stats[2 * c + 1] = ssq[0] / (float)n - m * m;
    }
}

__global__ void fn_bn_apply(float* __restrict__ x, const float* __restrict__ stats,
                            const float* __restrict__ g, const float* __restrict__ b, int n)
{
    int gid = blockIdx.x * blockDim.x + threadIdx.x;
    if (gid >= n * HID) return;
    int c = gid % HID;
    float m = stats[2 * c], v = stats[2 * c + 1];
    x[gid] = (x[gid] - m) * rsqrtf(v + EPSBN) * g[c] + b[c];
}

__global__ void fn_score(const float* __restrict__ f, const float* __restrict__ w,
                         float* __restrict__ s, int n)
{
    int v = blockIdx.x * blockDim.x + threadIdx.x;
    if (v >= n) return;
    float acc = 0.0f;
#pragma unroll 8
    for (int c = 0; c < HID; ++c) acc += f[(size_t)v * HID + c] * w[c];
    s[v] = 1.0f / (1.0f + expf(-acc));
}

// keep[i] = 1 iff #{j: s_j > s_i or (s_j==s_i and j<i)} < P/2  (top_k tie rule)
__global__ void fn_rank(const float* __restrict__ score, int* __restrict__ keep, int P)
{
    __shared__ float ss[4096];
    const int inst = blockIdx.x;
    const float* sb = score + (size_t)inst * P;
    const int i = blockIdx.y * blockDim.x + threadIdx.x;
    const float si = sb[i];
    int cnt = 0;
    for (int c0 = 0; c0 < P; c0 += 4096) {
        int m = min(4096, P - c0);
        __syncthreads();
        for (int t = threadIdx.x; t < m; t += blockDim.x) ss[t] = sb[c0 + t];
        __syncthreads();
        for (int jj = 0; jj < m; ++jj) {
            float sj = ss[jj];
            int j = c0 + jj;
            cnt += (sj > si) || (sj == si && j < i);
        }
    }
    keep[(size_t)inst * P + i] = (cnt < (P >> 1)) ? 1 : 0;
}

// Block-per-instance compaction (index order => ascending, matches sort(top)).
__global__ __launch_bounds__(1024) void fn_compact(const int* __restrict__ keep,
                                                   int* __restrict__ up, int P)
{
    __shared__ int flags[1024];
    __shared__ int wbase[33];
    const int inst = blockIdx.x;
    const int* kb = keep + (size_t)inst * P;
    const int tid = threadIdx.x, wid = tid >> 5;
    int base = 0;
    for (int c0 = 0; c0 < P; c0 += 1024) {
        int i = c0 + tid;
        int k = (i < P) ? kb[i] : 0;
        flags[tid] = k;
        __syncthreads();
        int pw = 0;
        for (int t = wid * 32; t < tid; ++t) pw += flags[t];   // intra-wave prefix
        if (tid == 0) {
            int acc = 0;
            for (int w = 0; w < 32; ++w) {
                int s = 0;
                for (int t = 0; t < 32; ++t) s += flags[w * 32 + t];
                wbase[w] = acc; acc += s;
            }
            wbase[32] = acc;
        }
        __syncthreads();
        if (k) up[(size_t)inst * (P >> 1) + base + wbase[wid] + pw] = inst * P + i;
        base += wbase[32];
        __syncthreads();
    }
}

__global__ void fn_gather_rows(const float* __restrict__ src, const int* __restrict__ idx,
                               float* __restrict__ dst, int n, int cols)
{
    int gid = blockIdx.x * blockDim.x + threadIdx.x;
    if (gid >= n * cols) return;
    int r = gid / cols, c = gid % cols;
    dst[gid] = src[(size_t)idx[r] * cols + c];
}

__global__ void fn_scatter_rows(const float* __restrict__ src, const int* __restrict__ idx,
                                float* __restrict__ dst, int n, int cols)
{
    int gid = blockIdx.x * blockDim.x + threadIdx.x;
    if (gid >= n * cols) return;
    int r = gid / cols, c = gid % cols;
    dst[(size_t)idx[r] * cols + c] = src[gid];
}

// concat16[:, coloff:coloff+80] = (f16)src   (concat16 is [N0, 256] f16)
__global__ void fn_copy_block_f16(const float* __restrict__ src,
                                  _Float16* __restrict__ dst, int coloff)
{
    int gid = blockIdx.x * blockDim.x + threadIdx.x;
    if (gid >= N0 * HID) return;
    int r = gid / HID, c = gid % HID;
    dst[(size_t)r * 256 + coloff + c] = (_Float16)src[gid];
}

__global__ void fn_zero_pad_concat(_Float16* __restrict__ dst)
{
    int gid = blockIdx.x * blockDim.x + threadIdx.x;
    if (gid >= N0 * 16) return;
    int r = gid / 16, c = gid % 16;
    dst[(size_t)r * 256 + 240 + c] = (_Float16)0.0f;
}

// ---------------------------------------------------------------------------
extern "C" void kernel_launch(void* const* d_in, const int* in_sizes, int n_in,
                              void* d_out, int out_size, void* d_ws, size_t ws_size,
                              hipStream_t stream)
{
    (void)in_sizes; (void)n_in; (void)out_size; (void)ws_size;
    const float* coord   = (const float*)d_in[0];
    const float* feat    = (const float*)d_in[1];
    const float* W_embed = (const float*)d_in[2];
    const float* b_embed = (const float*)d_in[3];
    const float* bn1_g   = (const float*)d_in[4];
    const float* bn1_b   = (const float*)d_in[5];
    const float* W_l     = (const float*)d_in[6];
    const float* b_l     = (const float*)d_in[7];
    const float* bn_g    = (const float*)d_in[8];
    const float* bn_b    = (const float*)d_in[9];
    const float* w_score = (const float*)d_in[10];
    const float* W0 = (const float*)d_in[11];
    const float* b0 = (const float*)d_in[12];
    const float* W1 = (const float*)d_in[13];
    const float* b1 = (const float*)d_in[14];
    const float* W2 = (const float*)d_in[15];
    const float* b2 = (const float*)d_in[16];
    float* out = (float*)d_out;

    const int Nl[3] = {65536, 32768, 16384};
    const int Pl[3] = {8192, 4096, 2048};

    // ---- workspace arena ----
    char* wp = (char*)d_ws;
    auto alloc = [&](size_t nbytes) -> char* {
        char* p = wp;
        wp += (nbytes + 255) & ~(size_t)255;
        return p;
    };
    float*     h      = (float*)alloc((size_t)N0 * HID * 4);    // current features
    _Float16*  X16    = (_Float16*)alloc((size_t)N0 * 96 * 2);  // padded f16 GEMM A
    float*     tmp    = (float*)alloc((size_t)N0 * HID * 4);    // push-down scratch
    float* featL[3];
    for (int l = 0; l < 3; ++l) featL[l] = (float*)alloc((size_t)Nl[l] * HID * 4);
    float* lat[3];
    for (int l = 0; l < 3; ++l) lat[l] = (float*)alloc((size_t)Nl[l] * HID * 4);
    _Float16* concat16 = (_Float16*)alloc((size_t)N0 * 256 * 2);
    float*    score    = (float*)alloc((size_t)N0 * 4);
    float*    stats    = (float*)alloc(256 * 4);
    float*    c1       = (float*)alloc((size_t)Nl[1] * 3 * 4);
    float*    c2       = (float*)alloc((size_t)Nl[2] * 3 * 4);
    int* knnIdx[3];
    for (int l = 0; l < 3; ++l) knnIdx[l] = (int*)alloc((size_t)Nl[l] * 3 * 4);
    int* up[2];
    for (int l = 0; l < 2; ++l) up[l] = (int*)alloc((size_t)Nl[l + 1] * 4);
    int* keep = (int*)alloc((size_t)N0 * 4);
    // f16 transposed weights (zero padded)
    _Float16* wtE = (_Float16*)alloc((size_t)80 * 32 * 2);     // embed: Ndpad=80, Kpad=32
    _Float16* wtL[3];
    for (int l = 0; l < 3; ++l) wtL[l] = (_Float16*)alloc((size_t)80 * 96 * 2);
    _Float16* wt0 = (_Float16*)alloc((size_t)128 * 256 * 2);
    _Float16* wt1 = (_Float16*)alloc((size_t)64 * 128 * 2);
    _Float16* wt2 = (_Float16*)alloc((size_t)16 * 64 * 2);
    // readout f16 temporaries aliased over dead f32 scratch
    _Float16* t1h = (_Float16*)tmp;   // N0*128*2 = 16.8MB <= 21MB
    _Float16* t2h = (_Float16*)h;     // N0*64*2  =  8.4MB <= 21MB

    auto nblk = [](int elems) { return (elems + 255) / 256; };
    auto gemm = [&](const _Float16* A, int Kpad, const _Float16* Wt,
                    const float* bias, const float* res, int ldr,
                    float* Cf, int ldc, _Float16* Ch, int ldch, int M, int Nd, int relu) {
        dim3 g(M / 16);
        switch ((Nd + 15) / 16) {
#define GEMM_CASE(NT) \
            case NT: fn_gemm_wmma<NT><<<g, 32, 0, stream>>>( \
                A, Kpad, Wt, Kpad, bias, res, ldr, Cf, ldc, Ch, ldch, Kpad, Nd, relu); break;
            GEMM_CASE(1) GEMM_CASE(2) GEMM_CASE(3) GEMM_CASE(4)
            GEMM_CASE(5) GEMM_CASE(6) GEMM_CASE(7) GEMM_CASE(8)
#undef GEMM_CASE
        }
    };

    // ---- convert weights to transposed padded f16 ----
    fn_cvt_wt16<<<nblk(80 * 32), 256, 0, stream>>>(W_embed, 6, 80, wtE, 32, 80);
    for (int l = 0; l < 3; ++l)
        fn_cvt_wt16<<<nblk(80 * 96), 256, 0, stream>>>(W_l + (size_t)l * HID * HID,
                                                       80, 80, wtL[l], 96, 80);
    fn_cvt_wt16<<<nblk(128 * 256), 256, 0, stream>>>(W0, 240, 120, wt0, 256, 128);
    fn_cvt_wt16<<<nblk(64 * 128), 256, 0, stream>>>(W1, 120, 60, wt1, 128, 64);
    fn_cvt_wt16<<<nblk(16 * 64), 256, 0, stream>>>(W2, 60, 13, wt2, 64, 16);
    fn_zero_pad_concat<<<nblk(N0 * 16), 256, 0, stream>>>(concat16);

    // ---- embedding + BN1 ----
    fn_cvt_a16<<<nblk(N0 * 32), 256, 0, stream>>>(feat, 6, X16, 32, N0);
    gemm(X16, 32, wtE, b_embed, nullptr, 0, h, HID, nullptr, 0, N0, 80, 0);
    fn_bn_stats<<<HID, 256, 0, stream>>>(h, stats, N0);
    fn_bn_apply<<<nblk(N0 * HID), 256, 0, stream>>>(h, stats, bn1_g, bn1_b, N0);

    const float* c_cur = coord;
    for (int l = 0; l < 3; ++l) {
        const int n = Nl[l], P = Pl[l];
        fn_knn<<<dim3(NB, P / 256), 256, 0, stream>>>(c_cur, knnIdx[l], P);
        // X = (f16)(h + sum_k h[knn]),  knn includes self
        fn_gather_sum_f16<<<nblk(n * 96), 256, 0, stream>>>(h, knnIdx[l], X16, n);
        // features = BN( h + relu(X @ W_l + b_l) )
        gemm(X16, 96, wtL[l], b_l + l * HID, h, HID, featL[l], HID, nullptr, 0, n, 80, 1);
        fn_bn_stats<<<HID, 256, 0, stream>>>(featL[l], stats, n);
        fn_bn_apply<<<nblk(n * HID), 256, 0, stream>>>(featL[l], stats,
                                                       bn_g + l * HID, bn_b + l * HID, n);
        if (l < 2) {
            fn_score<<<nblk(n), 256, 0, stream>>>(featL[l], w_score + l * HID, score, n);
            fn_rank<<<dim3(NB, P / 256), 256, 0, stream>>>(score, keep, P);
            fn_compact<<<NB, 1024, 0, stream>>>(keep, up[l], P);
        }
        // ---- push-down cascade (mirrors reference latest[] bookkeeping) ----
        const float* src = featL[l];
        for (int it = 0; it <= l; ++it) {
            int lvl = l - it;
            fn_gather_sum<<<nblk(Nl[lvl] * HID), 256, 0, stream>>>(src, knnIdx[lvl],
                                                                   tmp, Nl[lvl]);
            hipMemcpyAsync(lat[lvl], tmp, (size_t)Nl[lvl] * HID * 4,
                           hipMemcpyDeviceToDevice, stream);
            if (lvl > 0) {
                fn_scatter_rows<<<nblk(Nl[lvl] * HID), 256, 0, stream>>>(
                    tmp, up[lvl - 1], lat[lvl - 1], Nl[lvl], HID);
                src = lat[lvl - 1];
            }
        }
        fn_copy_block_f16<<<nblk(N0 * HID), 256, 0, stream>>>(tmp, concat16, l * HID);
        if (l < 2) {
            fn_gather_rows<<<nblk((n / 2) * HID), 256, 0, stream>>>(featL[l], up[l],
                                                                    h, n / 2, HID);
            float* cn = (l == 0) ? c1 : c2;
            fn_gather_rows<<<nblk((n / 2) * 3), 256, 0, stream>>>(c_cur, up[l],
                                                                  cn, n / 2, 3);
            c_cur = cn;
        }
    }

    // ---- MLP readout (f16 chained through t1h/t2h) ----
    gemm(concat16, 256, wt0, b0, nullptr, 0, nullptr, 0, t1h, 128, N0, 120, 1);
    gemm(t1h, 128, wt1, b1, nullptr, 0, nullptr, 0, t2h, 64, N0, 60, 1);
    gemm(t2h, 64, wt2, b2, nullptr, 0, out, 13, nullptr, 0, N0, 13, 0);
}
